// ADDerWithReconstruction_55937654063702
// MI455X (gfx1250) — compile-verified, matrix operand order checked by reading, not run on hardware
//
#include <hip/hip_runtime.h>
#include <hip/hip_bf16.h>

typedef __bf16  bf16_t;
typedef bf16_t  v16bf __attribute__((ext_vector_type(16)));
typedef float   v8f   __attribute__((ext_vector_type(8)));

#define D_FEAT 128

// ---------------------------------------------------------------------------
// Scatter: one wave per edge; lane l moves floats [4l,4l+4). Gather is a
// coalesced 512B read per wave; scatter is f32 atomics into agg[] which is
// L2-resident (25.6MB << 192MB L2).
// ---------------------------------------------------------------------------
__global__ __launch_bounds__(256)
void sage_scatter(const float* __restrict__ feat,
                  const int*   __restrict__ src,
                  const int*   __restrict__ dst,
                  float* __restrict__ agg,
                  float* __restrict__ cnt,
                  int E, int addCnt)
{
    int t = blockIdx.x * blockDim.x + threadIdx.x;
    int e = t >> 5;
    if (e >= E) return;
    int lane = t & 31;
    int s = src[e], d = dst[e];
    const float4 v = *reinterpret_cast<const float4*>(feat + (size_t)s * D_FEAT + lane * 4);
    float* p = agg + (size_t)d * D_FEAT + lane * 4;
    atomicAdd(p + 0, v.x);
    atomicAdd(p + 1, v.y);
    atomicAdd(p + 2, v.z);
    atomicAdd(p + 3, v.w);
    if (addCnt && lane == 0) atomicAdd(cnt + d, 1.0f);
}

// ---------------------------------------------------------------------------
// Pre-swizzle a 128x128 f32 weight matrix into bf16 WMMA B-fragment order:
//   Wsw[((colgroup*4 + kstep)*32 + lane)*16 + e]
// so each lane's B fragment is one contiguous 32-byte load.
// Fragment layout per ISA 7.12.2 (16-bit B, 32x16): lane = col(0..15) +
// 16*Khi, elements 0..7 -> K=khalf+e, 8..15 -> K=khalf+16+(e-8).
// ---------------------------------------------------------------------------
__global__ __launch_bounds__(256)
void pack_weight(const float* __restrict__ W, bf16_t* __restrict__ Wsw)
{
    int t = blockIdx.x * blockDim.x + threadIdx.x;   // 16384 total
    if (t >= D_FEAT * D_FEAT) return;
    int e  = t & 15;
    int l  = (t >> 4) & 31;
    int s  = (t >> 9) & 3;
    int c  = t >> 11;                                // colgroup 0..7
    int col   = c * 16 + (l & 15);
    int khalf = (l >> 4) * 8;
    int k     = s * 32 + khalf + ((e < 8) ? e : (e + 8));
    Wsw[t] = (bf16_t)W[k * D_FEAT + col];
}

__device__ __forceinline__ v16bf pack_a(const float* __restrict__ row, int k0)
{
    const float4 a0 = *reinterpret_cast<const float4*>(row + k0);
    const float4 a1 = *reinterpret_cast<const float4*>(row + k0 + 4);
    const float4 a2 = *reinterpret_cast<const float4*>(row + k0 + 16);
    const float4 a3 = *reinterpret_cast<const float4*>(row + k0 + 20);
    v16bf f;
    f[0]  = (bf16_t)a0.x; f[1]  = (bf16_t)a0.y; f[2]  = (bf16_t)a0.z; f[3]  = (bf16_t)a0.w;
    f[4]  = (bf16_t)a1.x; f[5]  = (bf16_t)a1.y; f[6]  = (bf16_t)a1.z; f[7]  = (bf16_t)a1.w;
    f[8]  = (bf16_t)a2.x; f[9]  = (bf16_t)a2.y; f[10] = (bf16_t)a2.z; f[11] = (bf16_t)a2.w;
    f[12] = (bf16_t)a3.x; f[13] = (bf16_t)a3.y; f[14] = (bf16_t)a3.z; f[15] = (bf16_t)a3.w;
    return f;
}

// ---------------------------------------------------------------------------
// Fused SAGEConv: out = diag(1/cnt)*(agg @ Wl) + self @ Wr + b  (+relu/drop).
// 8 waves/block; block owns a 16-row band, wave w owns columns [16w,16w+16).
// K=128 -> 4 k-steps, 2x v_wmma_f32_16x16x32_bf16 per step (8 WMMA / wave).
// Mean division is hoisted to the epilogue via a second accumulator.
// ---------------------------------------------------------------------------
template<bool RELU_DROP>
__global__ __launch_bounds__(256)
void sage_gemm(const float*  __restrict__ self_feat,
               const float*  __restrict__ agg,
               const float*  __restrict__ cnt,
               const bf16_t* __restrict__ WlSw,   // packed [8][4][32][16]
               const float*  __restrict__ bl,
               const bf16_t* __restrict__ WrSw,   // packed
               const float*  __restrict__ mask,
               float* __restrict__ out, int N)
{
    const int wave = threadIdx.x >> 5;
    const int lane = threadIdx.x & 31;
    const int row_base = blockIdx.x * 16;
    const int col   = wave * 16 + (lane & 15);
    const int idx   = lane & 15;
    const int khalf = (lane >> 4) * 8;

    int arow = row_base + idx;
    if (arow >= N) arow = N - 1;                 // clamp: no divergent loads
    const float* aggRow  = agg       + (size_t)arow * D_FEAT;
    const float* selfRow = self_feat + (size_t)arow * D_FEAT;
    const bf16_t* wl = WlSw + ((size_t)wave * 4 * 32) * 16 + (size_t)lane * 16;
    const bf16_t* wr = WrSw + ((size_t)wave * 4 * 32) * 16 + (size_t)lane * 16;

    v8f accL = {};   // agg  @ Wl (scaled by 1/cnt in epilogue)
    v8f accR = {};   // self @ Wr
    #pragma unroll
    for (int s4 = 0; s4 < 4; ++s4) {
        const int k0 = s4 * 32 + khalf;
        const v16bf aL = pack_a(aggRow,  k0);
        const v16bf aR = pack_a(selfRow, k0);
        const v16bf bL = *reinterpret_cast<const v16bf*>(wl + (size_t)s4 * 32 * 16);
        const v16bf bR = *reinterpret_cast<const v16bf*>(wr + (size_t)s4 * 32 * 16);
        accL = __builtin_amdgcn_wmma_f32_16x16x32_bf16(
                   false, aL, false, bL, (short)0, accL, false, false);
        accR = __builtin_amdgcn_wmma_f32_16x16x32_bf16(
                   false, aR, false, bR, (short)0, accR, false, false);
    }

    // C layout: VGPR r -> row r (lanes 0-15), row r+8 (lanes 16-31).
    const float bias = bl[col];
    const int   rofs = (lane >> 4) * 8;
    #pragma unroll
    for (int r = 0; r < 8; ++r) {
        const int row = row_base + r + rofs;
        if (row >= N) continue;
        const float inv = 1.0f / fmaxf(cnt[row], 1.0f);   // broadcast load
        float v = accL[r] * inv + accR[r] + bias;
        if (RELU_DROP) {
            v = fmaxf(v, 0.0f);
            v = v * mask[(size_t)row * D_FEAT + col] * 2.0f;
        }
        out[(size_t)row * D_FEAT + col] = v;
    }
}

extern "C" void kernel_launch(void* const* d_in, const int* in_sizes, int n_in,
                              void* d_out, int out_size, void* d_ws, size_t ws_size,
                              hipStream_t stream)
{
    const float* x    = (const float*)d_in[0];
    const int*   ei   = (const int*)  d_in[1];   // [2,E] int32
    const float* mask = (const float*)d_in[2];
    const float* W1l  = (const float*)d_in[3];
    const float* b1l  = (const float*)d_in[4];
    const float* W1r  = (const float*)d_in[5];
    const float* W2l  = (const float*)d_in[6];
    const float* b2l  = (const float*)d_in[7];
    const float* W2r  = (const float*)d_in[8];

    const int N = in_sizes[0] / D_FEAT;
    const int E = in_sizes[1] / 2;
    const int* src = ei;
    const int* dst = ei + E;

    const size_t nd = (size_t)N * D_FEAT;
    float*  agg = (float*)d_ws;                        // [N,128]
    float*  h   = agg + nd;                            // [N,128]
    float*  cnt = h + nd;                              // [N]
    size_t  cntPad = ((size_t)N + 63) & ~(size_t)63;   // keep 32B+ alignment
    bf16_t* w1lSw = (bf16_t*)(cnt + cntPad);           // 4 packed weight mats
    bf16_t* w1rSw = w1lSw + D_FEAT * D_FEAT;
    bf16_t* w2lSw = w1rSw + D_FEAT * D_FEAT;
    bf16_t* w2rSw = w2lSw + D_FEAT * D_FEAT;

    const int packBlocks  = (D_FEAT * D_FEAT + 255) / 256;
    const int scatBlocks  = (int)(((size_t)E * 32 + 255) / 256);
    const int gemmBlocks  = (N + 15) / 16;

    // Pre-swizzle all weights into bf16 WMMA fragment order (tiny, 128KB total).
    pack_weight<<<packBlocks, 256, 0, stream>>>(W1l, w1lSw);
    pack_weight<<<packBlocks, 256, 0, stream>>>(W1r, w1rSw);
    pack_weight<<<packBlocks, 256, 0, stream>>>(W2l, w2lSw);
    pack_weight<<<packBlocks, 256, 0, stream>>>(W2r, w2rSw);

    // ---- Layer 1 ----
    hipMemsetAsync(agg, 0, nd * sizeof(float), stream);
    hipMemsetAsync(cnt, 0, (size_t)N * sizeof(float), stream);
    sage_scatter<<<scatBlocks, 256, 0, stream>>>(x, src, dst, agg, cnt, E, 1);
    sage_gemm<true><<<gemmBlocks, 256, 0, stream>>>(x, agg, cnt, w1lSw, b1l,
                                                    w1rSw, mask, h, N);

    // ---- Layer 2 (same degrees; reuse cnt, re-zero agg) ----
    hipMemsetAsync(agg, 0, nd * sizeof(float), stream);
    sage_scatter<<<scatBlocks, 256, 0, stream>>>(h, src, dst, agg, cnt, E, 0);
    sage_gemm<false><<<gemmBlocks, 256, 0, stream>>>(h, agg, cnt, w2lSw, b2l,
                                                     w2rSw, nullptr,
                                                     (float*)d_out, N);
}
